// TransformerEncoderBlock_31069793419266
// MI455X (gfx1250) — matrix operand order repaired and checked by reading, not compiled, not run on hardware
//
#include <hip/hip_runtime.h>
#include <hip/hip_bf16.h>
#include <math.h>

// ---------------------------------------------------------------------------
// CDNA5 / gfx1250 transformer encoder block (band window = 1 attention).
// bf16 WMMA (v_wmma_f32_16x16x32_bf16) for all GEMMs, f32 accumulate.
// GEMM staging uses GLOBAL_LOAD_ASYNC_TO_LDS_B128 (ASYNCcnt) + double-buffered
// LDS; wave tile 64x32 = 8 WMMAs per K-step.
// ---------------------------------------------------------------------------

typedef __bf16 bf16_t;
typedef __attribute__((ext_vector_type(16))) __bf16 v16bf;
typedef __attribute__((ext_vector_type(8)))  __bf16 v8bf;
typedef __attribute__((ext_vector_type(8)))  float   v8f;

namespace {
constexpr int Bc = 4, Sc = 2048, Dc = 512, Hc = 8, Kc = 512, Fc = 2048;
constexpr int BSc  = Bc * Sc;          // 8192 rows
constexpr int HKc  = Hc * Kc;          // 4096
constexpr int NQKV = 3 * HKc;          // 12288 fused qkv columns
constexpr float ATTN_SCALE = 0.04419417382415922f;  // 1/sqrt(512)
}

union FragBF { v16bf v; v8bf h[2]; };

// Low 32 bits of a generic LDS address == wave-relative LDS byte offset
// (ISA 10.2: generic->LDS mapping uses addr[31:0]; HW adds LDS_BASE).
__device__ __forceinline__ unsigned lds_off(const void* p) {
  return (unsigned)(size_t)p;
}

__device__ __forceinline__ void async_ld_b128(unsigned lds, const void* g) {
  asm volatile("global_load_async_to_lds_b128 %0, %1, off"
               :: "v"(lds), "v"((unsigned long long)(size_t)g)
               : "memory");
}

__device__ __forceinline__ void wait_async0() {
  asm volatile("s_wait_asynccnt 0x0" ::: "memory");
}

// ---------------------------------------------------------------------------
// Weight prep: fused qkv weight, transposed to (n_global, d) row-major bf16.
// n_global in [0,12288): 0..4095 = Q(h*512+k), 4096.. = K, 8192.. = V.
// Also builds the concatenated qkv bias (f32).
// ---------------------------------------------------------------------------
__global__ __launch_bounds__(256) void k_build_qkvT(
    bf16_t* __restrict__ dst, float* __restrict__ bias_dst,
    const float* __restrict__ Wq, const float* __restrict__ Wk,
    const float* __restrict__ Wv, const float* __restrict__ bq,
    const float* __restrict__ bk, const float* __restrict__ bv) {
  int i = blockIdx.x * 256 + threadIdx.x;          // i = ng*512 + d
  if (i >= NQKV * Dc) return;
  int d  = i & 511;
  int ng = i >> 9;
  int sel = ng >> 12;                              // 0=q,1=k,2=v
  int n   = ng & 4095;                             // h*512 + kk
  const float* W = (sel == 0) ? Wq : (sel == 1) ? Wk : Wv;
  dst[i] = (bf16_t)W[(size_t)d * HKc + n];         // W is (D, H*K) row-major
  if (i < NQKV) {
    int s2 = i >> 12, nn = i & 4095;
    bias_dst[i] = (s2 == 0) ? bq[nn] : (s2 == 1) ? bk[nn] : bv[nn];
  }
}

// Generic transpose + f32->bf16: src is R x C row-major, dst is C x R.
__global__ __launch_bounds__(256) void k_transpose_cvt(
    bf16_t* __restrict__ dst, const float* __restrict__ src, int R, int C) {
  int i = blockIdx.x * 256 + threadIdx.x;          // i = c*R + r
  if (i >= R * C) return;
  int r = i % R, c = i / R;
  dst[i] = (bf16_t)src[(size_t)r * C + c];
}

__global__ __launch_bounds__(256) void k_cvt_bf16(
    bf16_t* __restrict__ dst, const float* __restrict__ src, int n) {
  int i = blockIdx.x * 256 + threadIdx.x;
  if (i < n) dst[i] = (bf16_t)src[i];
}

// ---------------------------------------------------------------------------
// WMMA GEMM: C[M,N] = A[M,Kd] * BT[N,Kd]^T (+bias) (+gelu) -> f32 and/or bf16.
// 256 threads = 8 waves; block tile 128x128; wave tile 64x32 (4x2 WMMA accs).
// Double-buffered LDS filled by global_load_async_to_lds_b128.
// ---------------------------------------------------------------------------
__global__ __launch_bounds__(256) void k_gemm_bf16(
    const bf16_t* __restrict__ A, const bf16_t* __restrict__ BT,
    const float* __restrict__ bias, float* __restrict__ Cf,
    bf16_t* __restrict__ Cb, int M, int N, int Kd, int act) {
  __shared__ bf16_t sA[2][128][40];  // 128 x 32 halves (+8 pad), x2 buffers
  __shared__ bf16_t sB[2][128][40];

  const int tid  = threadIdx.x;
  const int m0   = blockIdx.y * 128;
  const int n0   = blockIdx.x * 128;
  const int wv   = tid >> 5;
  const int lane = tid & 31;
  const int lr   = lane & 15;       // row/col within 16-tile
  const int lh   = lane >> 4;       // half-wave select
  const int wm   = (wv & 1) * 64;   // 2 waves along M
  const int wn   = (wv >> 1) * 32;  // 4 waves along N

  // Staging: 1024 16B-chunks per K-step (A 512 + B 512), 4 per thread.
  const int srow = tid >> 2;             // 0..63
  const int scol = (tid & 3) * 8;        // half index 0/8/16/24
  const bf16_t* gA0 = A  + (size_t)(m0 + srow) * Kd + scol;
  const bf16_t* gA1 = gA0 + (size_t)64 * Kd;
  const bf16_t* gB0 = BT + (size_t)(n0 + srow) * Kd + scol;
  const bf16_t* gB1 = gB0 + (size_t)64 * Kd;

  auto stage = [&](int buf, int kt) {
    async_ld_b128(lds_off(&sA[buf][srow][scol]),      gA0 + kt);
    async_ld_b128(lds_off(&sA[buf][64 + srow][scol]), gA1 + kt);
    async_ld_b128(lds_off(&sB[buf][srow][scol]),      gB0 + kt);
    async_ld_b128(lds_off(&sB[buf][64 + srow][scol]), gB1 + kt);
  };

  v8f acc[4][2];
  const v8f vz = {0.f, 0.f, 0.f, 0.f, 0.f, 0.f, 0.f, 0.f};
#pragma unroll
  for (int i = 0; i < 4; ++i)
#pragma unroll
    for (int j = 0; j < 2; ++j) acc[i][j] = vz;

  stage(0, 0);
  int buf = 0;
  for (int kt = 0; kt < Kd; kt += 32) {
    wait_async0();        // this wave's fills for `buf` have landed
    __syncthreads();      // everyone's fills for `buf` have landed
    if (kt + 32 < Kd) stage(buf ^ 1, kt + 32);  // prefetch next K-step

    // A fragment (ISA 16-bit A 16x32 layout): lane holds row M=lr;
    // halves 0-7 = K[lh*8 .. +7], halves 8-15 = K[16+lh*8 .. +7].
    FragBF fa[4], fb[2];
#pragma unroll
    for (int i = 0; i < 4; ++i) {
      int r = wm + i * 16 + lr;
      fa[i].h[0] = *(const v8bf*)&sA[buf][r][lh * 8];
      fa[i].h[1] = *(const v8bf*)&sA[buf][r][16 + lh * 8];
    }
    // B fragment: lane holds col N=lr; same K layout as A.
#pragma unroll
    for (int j = 0; j < 2; ++j) {
      int r = wn + j * 16 + lr;
      fb[j].h[0] = *(const v8bf*)&sB[buf][r][lh * 8];
      fb[j].h[1] = *(const v8bf*)&sB[buf][r][16 + lh * 8];
    }
#pragma unroll
    for (int i = 0; i < 4; ++i)
#pragma unroll
      for (int j = 0; j < 2; ++j)
        acc[i][j] = __builtin_amdgcn_wmma_f32_16x16x32_bf16(
            false, fa[i].v, false, fb[j].v, (short)0, acc[i][j], false, false);
    buf ^= 1;
    // No trailing barrier needed: each wave's LDS reads complete (dscnt wait
    // before WMMA) before it signals the next iteration's barrier.
  }

  // Epilogue. C layout: VGPR v -> (M = v + 8*lh, N = lr) within each 16x16.
#pragma unroll
  for (int i = 0; i < 4; ++i)
#pragma unroll
    for (int j = 0; j < 2; ++j) {
      int col = n0 + wn + j * 16 + lr;
      float bv = bias ? bias[col] : 0.f;
#pragma unroll
      for (int v = 0; v < 8; ++v) {
        int row = m0 + wm + i * 16 + v + 8 * lh;
        float val = acc[i][j][v] + bv;
        if (act == 1) val = 0.5f * val * (1.f + erff(val * 0.70710678118f));
        size_t o = (size_t)row * N + col;
        if (Cf) Cf[o] = val;
        if (Cb) Cb[o] = (bf16_t)val;
      }
    }
}

// ---------------------------------------------------------------------------
// Banded attention (window 1): one wave per (b,s,h) row. 3 dots of length 512,
// 3-way softmax, weighted V combine. qkv layout: [BS][12288] bf16.
// ---------------------------------------------------------------------------
__global__ __launch_bounds__(256) void k_attn(
    const bf16_t* __restrict__ qkv, bf16_t* __restrict__ ctx) {
  const int w    = blockIdx.x * 8 + (threadIdx.x >> 5);  // row in [0, BS*H)
  const int lane = threadIdx.x & 31;
  const int h = w % Hc;
  const int t = w / Hc;
  const int s = t % Sc;
  const int b = t / Sc;
  const int e0 = lane * 16;

  const size_t qoff = (size_t)(b * Sc + s) * NQKV + (size_t)h * Kc + e0;
  float qv[16];
  {
    v8bf q0 = *(const v8bf*)(qkv + qoff);
    v8bf q1 = *(const v8bf*)(qkv + qoff + 8);
#pragma unroll
    for (int e = 0; e < 8; ++e) { qv[e] = (float)q0[e]; qv[8 + e] = (float)q1[e]; }
  }

  float sc[3];
  bool valid[3];
#pragma unroll
  for (int dlt = 0; dlt < 3; ++dlt) {
    int ss = s + dlt - 1;
    valid[dlt] = (ss >= 0) && (ss < Sc);
    float d = -1e9f;
    if (valid[dlt]) {  // wave-uniform branch
      const bf16_t* kp =
          qkv + (size_t)(b * Sc + ss) * NQKV + HKc + (size_t)h * Kc + e0;
      v8bf k0 = *(const v8bf*)kp;
      v8bf k1 = *(const v8bf*)(kp + 8);
      float a = 0.f;
#pragma unroll
      for (int e = 0; e < 8; ++e) {
        a += qv[e] * (float)k0[e];
        a += qv[8 + e] * (float)k1[e];
      }
#pragma unroll
      for (int off = 16; off > 0; off >>= 1) a += __shfl_xor(a, off, 32);
      d = a * ATTN_SCALE;
    }
    sc[dlt] = d;
  }

  float m  = fmaxf(sc[0], fmaxf(sc[1], sc[2]));
  float e0w = expf(sc[0] - m), e1w = expf(sc[1] - m), e2w = expf(sc[2] - m);
  float inv = 1.f / (e0w + e1w + e2w);
  float p[3] = {e0w * inv, e1w * inv, e2w * inv};

  float o[16];
#pragma unroll
  for (int e = 0; e < 16; ++e) o[e] = 0.f;
#pragma unroll
  for (int dlt = 0; dlt < 3; ++dlt) {
    if (!valid[dlt]) continue;
    int ss = s + dlt - 1;
    const bf16_t* vp =
        qkv + (size_t)(b * Sc + ss) * NQKV + 2 * HKc + (size_t)h * Kc + e0;
    v8bf v0 = *(const v8bf*)vp;
    v8bf v1 = *(const v8bf*)(vp + 8);
#pragma unroll
    for (int e = 0; e < 8; ++e) {
      o[e]     += p[dlt] * (float)v0[e];
      o[8 + e] += p[dlt] * (float)v1[e];
    }
  }

  bf16_t* cp = ctx + (size_t)(b * Sc + s) * HKc + (size_t)h * Kc + e0;
#pragma unroll
  for (int e = 0; e < 16; ++e) cp[e] = (bf16_t)o[e];
}

// ---------------------------------------------------------------------------
// Fused residual add + LayerNorm over D=512. One wave per row (16 elems/lane).
// Writes f32 output and (optionally) a bf16 copy for the next GEMM.
// ---------------------------------------------------------------------------
__global__ __launch_bounds__(256) void k_add_ln(
    const float* __restrict__ a, const float* __restrict__ r,
    const float* __restrict__ g, const float* __restrict__ be,
    float* __restrict__ of, bf16_t* __restrict__ ob) {
  const int row  = blockIdx.x * 8 + (threadIdx.x >> 5);
  const int lane = threadIdx.x & 31;
  const size_t base = (size_t)row * Dc + lane * 16;

  float x[16];
  float s1 = 0.f, s2 = 0.f;
#pragma unroll
  for (int e = 0; e < 16; ++e) {
    x[e] = a[base + e] + r[base + e];
    s1 += x[e];
    s2 += x[e] * x[e];
  }
#pragma unroll
  for (int off = 16; off > 0; off >>= 1) {
    s1 += __shfl_xor(s1, off, 32);
    s2 += __shfl_xor(s2, off, 32);
  }
  float mean = s1 * (1.f / 512.f);
  float var  = s2 * (1.f / 512.f) - mean * mean;
  float inv  = rsqrtf(var + 1e-3f);
#pragma unroll
  for (int e = 0; e < 16; ++e) {
    int c = lane * 16 + e;
    float y = (x[e] - mean) * inv * g[c] + be[c];
    of[base + e] = y;
    if (ob) ob[base + e] = (bf16_t)y;
  }
}

// ---------------------------------------------------------------------------
// Host-side orchestration.
// ---------------------------------------------------------------------------
extern "C" void kernel_launch(void* const* d_in, const int* in_sizes, int n_in,
                              void* d_out, int out_size, void* d_ws,
                              size_t ws_size, hipStream_t stream) {
  const float* x  = (const float*)d_in[0];
  const float* Wq = (const float*)d_in[1];
  const float* bq = (const float*)d_in[2];
  const float* Wk = (const float*)d_in[3];
  const float* bk = (const float*)d_in[4];
  const float* Wv = (const float*)d_in[5];
  const float* bv = (const float*)d_in[6];
  const float* Wo = (const float*)d_in[7];
  const float* bo = (const float*)d_in[8];
  const float* W1 = (const float*)d_in[9];
  const float* b1 = (const float*)d_in[10];
  const float* W2 = (const float*)d_in[11];
  const float* b2 = (const float*)d_in[12];
  const float* g1 = (const float*)d_in[13];
  const float* be1 = (const float*)d_in[14];
  const float* g2 = (const float*)d_in[15];
  const float* be2 = (const float*)d_in[16];

  char* ws = (char*)d_ws;
  size_t o = 0;
  auto alloc = [&](size_t bytes) -> char* {
    char* p = ws + o;
    o += (bytes + 255) & ~size_t(255);
    return p;
  };
  bf16_t* WqkvT = (bf16_t*)alloc((size_t)NQKV * Dc * 2);   // 12.6 MB
  bf16_t* WoT   = (bf16_t*)alloc((size_t)Dc * HKc * 2);    //  4.2 MB
  bf16_t* W1T   = (bf16_t*)alloc((size_t)Fc * Dc * 2);     //  2.1 MB
  bf16_t* W2T   = (bf16_t*)alloc((size_t)Dc * Fc * 2);     //  2.1 MB
  float*  bqkv  = (float*)alloc((size_t)NQKV * 4);
  bf16_t* xb    = (bf16_t*)alloc((size_t)BSc * Dc * 2);    //  8.4 MB
  bf16_t* qkvb  = (bf16_t*)alloc((size_t)BSc * NQKV * 2);  // 201 MB
  bf16_t* ctxb  = (bf16_t*)alloc((size_t)BSc * HKc * 2);   //  67 MB
  float*  attnf = (float*)alloc((size_t)BSc * Dc * 4);     //  17 MB
  float*  out1f = (float*)alloc((size_t)BSc * Dc * 4);
  bf16_t* out1b = (bf16_t*)alloc((size_t)BSc * Dc * 2);
  bf16_t* hbuf  = (bf16_t*)alloc((size_t)BSc * Fc * 2);    //  34 MB
  float*  ffnf  = (float*)alloc((size_t)BSc * Dc * 4);

  dim3 blk(256);
  // Weight prep (bf16, N-major so B fragments are contiguous in K).
  k_build_qkvT<<<(NQKV * Dc + 255) / 256, blk, 0, stream>>>(
      WqkvT, bqkv, Wq, Wk, Wv, bq, bk, bv);
  k_transpose_cvt<<<(HKc * Dc + 255) / 256, blk, 0, stream>>>(WoT, Wo, HKc, Dc);
  k_transpose_cvt<<<(Dc * Fc + 255) / 256, blk, 0, stream>>>(W1T, W1, Dc, Fc);
  k_transpose_cvt<<<(Fc * Dc + 255) / 256, blk, 0, stream>>>(W2T, W2, Fc, Dc);
  k_cvt_bf16<<<(BSc * Dc + 255) / 256, blk, 0, stream>>>(xb, x, BSc * Dc);

  // Fused QKV projection: 8192 x 12288 x 512 -> bf16 qkv.
  k_gemm_bf16<<<dim3(NQKV / 128, BSc / 128), blk, 0, stream>>>(
      xb, WqkvT, bqkv, nullptr, qkvb, BSc, NQKV, Dc, 0);
  // Banded attention -> ctx bf16.
  k_attn<<<(BSc * Hc) / 8, blk, 0, stream>>>(qkvb, ctxb);
  // Output projection: 8192 x 512 x 4096.
  k_gemm_bf16<<<dim3(Dc / 128, BSc / 128), blk, 0, stream>>>(
      ctxb, WoT, bo, attnf, nullptr, BSc, Dc, HKc, 0);
  // out1 = LN(x + attn)
  k_add_ln<<<BSc / 8, blk, 0, stream>>>(x, attnf, g1, be1, out1f, out1b);
  // FFN1 with exact gelu: 8192 x 2048 x 512 -> bf16 h.
  k_gemm_bf16<<<dim3(Fc / 128, BSc / 128), blk, 0, stream>>>(
      out1b, W1T, b1, nullptr, hbuf, BSc, Fc, Dc, 1);
  // FFN2: 8192 x 512 x 2048.
  k_gemm_bf16<<<dim3(Dc / 128, BSc / 128), blk, 0, stream>>>(
      hbuf, W2T, b2, ffnf, nullptr, BSc, Dc, Fc, 0);
  // out = LN(out1 + ffn)
  k_add_ln<<<BSc / 8, blk, 0, stream>>>(out1f, ffnf, g2, be2, (float*)d_out,
                                        nullptr);
}
